// FourLayer_64F_72138270704334
// MI455X (gfx1250) — compile-verified, matrix operand order checked by reading, not run on hardware
//
#include <hip/hip_runtime.h>
#include <math.h>

typedef __attribute__((ext_vector_type(16))) _Float16 v16h;
typedef __attribute__((ext_vector_type(8)))  _Float16 v8h;
typedef __attribute__((ext_vector_type(8)))  float    v8f;

#define NIMG 80
#define NEG_BIG (-3.0e38f)

// Load the two contiguous 8-half runs a lane needs for a 16-bit WMMA A/B
// operand (K-halves kb..kb+7 and kb+16..kb+23), as 2x b128 loads.
static __device__ __forceinline__ v16h ldpair(const _Float16* __restrict__ row, int kb) {
    v8h x = *(const v8h*)(row + kb);
    v8h y = *(const v8h*)(row + kb + 16);
    v16h r;
#pragma unroll
    for (int e = 0; e < 8; ++e) { r[e] = x[e]; r[e + 8] = y[e]; }
    return r;
}

static __device__ __forceinline__ v16h zero16() {
    v16h r;
#pragma unroll
    for (int e = 0; e < 16; ++e) r[e] = (_Float16)0.f;
    return r;
}

// ---------------------------------------------------------------------------
// Weight / BN prep.  Weights prepped TRANSPOSED: wkT[co][Kpad], so WMMA B
// operands are contiguous 16B runs per lane.
// ---------------------------------------------------------------------------
__global__ void prep_wk(const float* __restrict__ w, _Float16* __restrict__ wkT,
                        int Cin, int Kpad) {
    int idx = blockIdx.x * blockDim.x + threadIdx.x;
    if (idx >= 64 * Kpad) return;
    int co = idx / Kpad, k = idx % Kpad;
    int Kreal = Cin * 9;
    float v = (k < Kreal) ? w[(size_t)co * Kreal + k] : 0.f;   // w[co][ci][3][3]
    wkT[idx] = (_Float16)v;
}

__global__ void prep_bn(const float* g, const float* b, const float* m,
                        const float* v, float* scale, float* shift, int C) {
    int t = blockIdx.x * blockDim.x + threadIdx.x;
    if (t >= C) return;
    float s = g[t] / sqrtf(v[t] + 1e-5f);
    scale[t] = s;
    shift[t] = b[t] - m[t] * s;
}

__global__ void prep_bias(const float* b, float* scale, float* shift, int C) {
    int t = blockIdx.x * blockDim.x + threadIdx.x;
    if (t >= C) return;
    scale[t] = 1.f;
    shift[t] = b[t];
}

// ---------------------------------------------------------------------------
// conv0 (3->64, pad1) + BN + LReLU + 2x2 maxpool, scalar (Cin=3 is tiny)
// ---------------------------------------------------------------------------
__global__ void conv0_pool(const float* __restrict__ in1, const float* __restrict__ in2,
                           const float* __restrict__ w, const float* __restrict__ scale,
                           const float* __restrict__ shift, _Float16* __restrict__ y) {
    int idx = blockIdx.x * blockDim.x + threadIdx.x;
    int total = NIMG * 64 * 42 * 42;
    if (idx >= total) return;
    int ox = idx % 42; int t = idx / 42;
    int oy = t % 42;  t /= 42;
    int co = t % 64;  int n = t / 64;
    const float* xin = (n < 75) ? (in1 + (size_t)n * 3 * 84 * 84)
                                : (in2 + (size_t)(n - 75) * 3 * 84 * 84);
    const float* wc = w + (size_t)co * 27;
    float sc = scale[co], sh = shift[co];
    float best = NEG_BIG;
    for (int dy = 0; dy < 2; ++dy)
        for (int dx = 0; dx < 2; ++dx) {
            int py = oy * 2 + dy, px = ox * 2 + dx;
            float acc = 0.f;
            for (int ci = 0; ci < 3; ++ci)
                for (int r = 0; r < 3; ++r) {
                    int iy = py - 1 + r;
                    if (iy < 0 || iy >= 84) continue;
                    for (int s = 0; s < 3; ++s) {
                        int ix = px - 1 + s;
                        if (ix < 0 || ix >= 84) continue;
                        acc += xin[(ci * 84 + iy) * 84 + ix] * wc[ci * 9 + r * 3 + s];
                    }
                }
            float v = acc * sc + sh;
            v = v >= 0.f ? v : 0.2f * v;
            best = fmaxf(best, v);
        }
    y[idx] = (_Float16)best;
}

// ---------------------------------------------------------------------------
// Implicit-GEMM 3x3 conv via WMMA. M=16 spatial, N=64 (4 waves x 16), K=Cin*9.
// A k-table (per-k gather offset + (dy,dx)) is built once in LDS so the inner
// im2col builder has no div/mod; per-thread spatial bases and LDS destinations
// are hoisted (kk = tid&31 is loop-invariant).  A operand read as 2x
// ds_load_b128 (swizzled layout), B as 2x global b128 from transposed weights.
// Fused BN(+bias) + LeakyReLU, fp16 out, optional transposed copy.
// ---------------------------------------------------------------------------
__global__ __launch_bounds__(128)
void conv3x3_wmma(const _Float16* __restrict__ x, const _Float16* __restrict__ wkT,
                  const float* __restrict__ scale, const float* __restrict__ shift,
                  _Float16* __restrict__ y, _Float16* __restrict__ yT,
                  int IH, int IW, int OH, int OW, int Cin, int stride, int pad,
                  int nk, int Kreal) {
    const int P    = OH * OW;
    const int n    = blockIdx.y;
    const int p0   = blockIdx.x * 16;
    const int tid  = threadIdx.x;
    const int wv   = tid >> 5;
    const int lane = tid & 31;
    const int lhi  = lane >> 4;
    const int lrow = lane & 15;
    __shared__ __align__(32) _Float16 Aswz[32][16];   // [operand lane][element]
    __shared__ int ktab_off[576];                     // ci*IH*IW + dy*IW + dx
    __shared__ int ktab_d[576];                       // (dy<<16) | (dx & 0xffff)
    // ---- one-time k-table (all div/mod hoisted here) ----
    for (int k = tid; k < nk * 32; k += 128) {
        int ci = k / 9, tt = k % 9;
        int dy = tt / 3 - pad, dx = tt % 3 - pad;
        ktab_off[k] = ci * IH * IW + dy * IW + dx;
        ktab_d[k]   = (dy << 16) | (dx & 0xffff);
    }
    // ---- per-thread loop-invariants: fixed kk, 4 fixed rows ----
    const int kkf = tid & 31;
    const int wl  = ((kkf >> 3) & 1);             // which lane half this kk feeds
    const int kq  = kkf - 8 * wl;
    const int we  = (kq & 16) ? (kq - 8) : kq;    // element within lane slot
    int  pbase[4]; int oys[4]; int oxs[4]; bool pvld[4]; _Float16* dst[4];
#pragma unroll
    for (int it = 0; it < 4; ++it) {
        int r = (tid >> 5) + 4 * it;
        int p = p0 + r;
        pvld[it] = p < P;
        int pp = pvld[it] ? p : 0;
        int oy = pp / OW, ox = pp % OW;
        oys[it] = oy * stride;
        oxs[it] = ox * stride;
        pbase[it] = oys[it] * IW + oxs[it];
        dst[it] = &Aswz[r + 16 * wl][we];
    }
    v8f c = {};
    const _Float16* xin  = x + (size_t)n * Cin * IH * IW;
    const int Kpad = nk * 32;
    const _Float16* wrow = wkT + (size_t)(wv * 16 + lrow) * Kpad;
    const int kb = lhi ? 8 : 0;
    __syncthreads();   // k-table ready
    for (int ks = 0; ks < nk; ++ks) {
        int k0 = ks * 32;
        int k  = k0 + kkf;
        int koff = ktab_off[k];
        int kd   = ktab_d[k];
        int dy = kd >> 16;
        int dx = (int)(short)(kd & 0xffff);
        bool kok = k < Kreal;
#pragma unroll
        for (int it = 0; it < 4; ++it) {
            float val = 0.f;
            int iy = oys[it] + dy, ix = oxs[it] + dx;
            if (kok && pvld[it] && iy >= 0 && iy < IH && ix >= 0 && ix < IW)
                val = (float)xin[pbase[it] + koff];
            *dst[it] = (_Float16)val;
        }
        __syncthreads();
        v16h a = *(const v16h*)&Aswz[lane][0];
        v16h b = ldpair(wrow + k0, kb);
        c = __builtin_amdgcn_wmma_f32_16x16x32_f16(false, a, false, b,
                                                   (short)0, c, false, false);
        __syncthreads();  // WMMA operands consumed before tile rebuild
    }
#pragma unroll
    for (int r = 0; r < 8; ++r) {
        int m = r + 8 * lhi;
        int p = p0 + m;
        if (p >= P) continue;
        int ch = wv * 16 + lrow;
        float v = c[r] * scale[ch] + shift[ch];
        v = v >= 0.f ? v : 0.2f * v;
        y[((size_t)n * 64 + ch) * P + p] = (_Float16)v;
        if (yT) yT[((size_t)n * P + p) * 64 + ch] = (_Float16)v;
    }
}

// ---------------------------------------------------------------------------
// Generic max pool (fp16)
// ---------------------------------------------------------------------------
__global__ void maxpool(const _Float16* __restrict__ x, _Float16* __restrict__ y,
                        int C, int IH, int IW, int k, int s, int OH, int OW) {
    int idx = blockIdx.x * blockDim.x + threadIdx.x;
    int total = C * OH * OW;
    if (idx >= total) return;
    int ox = idx % OW; int oy = (idx / OW) % OH; int c = idx / (OH * OW);
    float best = NEG_BIG;
    const _Float16* xp = x + (size_t)c * IH * IW;
    for (int dy = 0; dy < k; ++dy)
        for (int dx = 0; dx < k; ++dx)
            best = fmaxf(best, (float)xp[(oy * s + dy) * IW + ox * s + dx]);
    y[idx] = (_Float16)best;
}

// ---------------------------------------------------------------------------
// Per-descriptor L2 norm over channels (from transposed features)
// ---------------------------------------------------------------------------
__global__ void norm_kernel(const _Float16* __restrict__ fT, float* __restrict__ nrm) {
    int idx = blockIdx.x * blockDim.x + threadIdx.x;
    if (idx >= NIMG * 441) return;
    const v8h* p = (const v8h*)(fT + (size_t)idx * 64);
    float acc = 0.f;
#pragma unroll
    for (int g = 0; g < 8; ++g) {
        v8h v = p[g];
#pragma unroll
        for (int e = 0; e < 8; ++e) { float f = (float)v[e]; acc += f * f; }
    }
    nrm[idx] = sqrtf(acc);
}

// ---------------------------------------------------------------------------
// Transposed convolutions (tiny), BN+LReLU fused where needed
// ---------------------------------------------------------------------------
__global__ void deconv0(const _Float16* __restrict__ h, const float* __restrict__ w,
                        const float* __restrict__ scale, const float* __restrict__ shift,
                        float* __restrict__ y) {
    int idx = blockIdx.x * blockDim.x + threadIdx.x;       // [N][64][4][4]
    if (idx >= NIMG * 64 * 16) return;
    int px = idx % 16; int co = (idx / 16) % 64; int n = idx / (64 * 16);
    float acc = 0.f;
    for (int ci = 0; ci < 64; ++ci)
        acc += (float)h[n * 64 + ci] * w[((size_t)ci * 64 + co) * 16 + px];
    float v = acc * scale[co] + shift[co];
    y[idx] = v >= 0.f ? v : 0.2f * v;
}

__global__ void deconv1(const float* __restrict__ x, const float* __restrict__ w,
                        const float* __restrict__ scale, const float* __restrict__ shift,
                        float* __restrict__ y) {
    int idx = blockIdx.x * blockDim.x + threadIdx.x;       // [N][64][10][10]
    if (idx >= NIMG * 64 * 100) return;
    int ox = idx % 10; int oy = (idx / 10) % 10; int co = (idx / 100) % 64;
    int n = idx / 6400;
    float acc = 0.f;
    for (int iy = 0; iy < 4; ++iy) {
        int ky = oy - 2 * iy; if (ky < 0 || ky >= 4) continue;
        for (int ix = 0; ix < 4; ++ix) {
            int kx = ox - 2 * ix; if (kx < 0 || kx >= 4) continue;
            const float* xp = x + (size_t)n * 64 * 16 + iy * 4 + ix;
            const float* wp = w + (size_t)co * 16 + ky * 4 + kx;
            for (int ci = 0; ci < 64; ++ci)
                acc += xp[(size_t)ci * 16] * wp[(size_t)ci * 64 * 16];
        }
    }
    float v = acc * scale[co] + shift[co];
    y[idx] = v >= 0.f ? v : 0.2f * v;
}

__global__ void deconv2(const float* __restrict__ x, const float* __restrict__ w,
                        const float* __restrict__ bias, float* __restrict__ y) {
    int idx = blockIdx.x * blockDim.x + threadIdx.x;       // [N][441]
    if (idx >= NIMG * 441) return;
    int ox = idx % 21; int oy = (idx / 21) % 21; int n = idx / 441;
    float acc = bias[0];
    for (int iy = 0; iy < 10; ++iy) {
        int ky = oy - 2 * iy; if (ky < 0 || ky >= 3) continue;
        for (int ix = 0; ix < 10; ++ix) {
            int kx = ox - 2 * ix; if (kx < 0 || kx >= 3) continue;
            const float* xp = x + ((size_t)n * 64) * 100 + iy * 10 + ix;
            const float* wp = w + ky * 3 + kx;
            for (int ci = 0; ci < 64; ++ci)
                acc += xp[(size_t)ci * 100] * wp[(size_t)ci * 9];
        }
    }
    y[idx] = acc;
}

// ---------------------------------------------------------------------------
// Per-image Gaussian mask + fold in 1/norm:  scl[n][p] = mask[n][p]/norm[n][p]
// ---------------------------------------------------------------------------
__global__ __launch_bounds__(64)
void mask_kernel(const _Float16* __restrict__ hb, const float* __restrict__ d2,
                 const float* __restrict__ nrm,
                 const float* __restrict__ sxw, const float* __restrict__ sxb,
                 const float* __restrict__ syw, const float* __restrict__ syb,
                 const float* __restrict__ thw, const float* __restrict__ thb,
                 float* __restrict__ scl) {
    const int n = blockIdx.x, t = threadIdx.x;
    __shared__ float feat[64];
    __shared__ float lin[3][6];
    __shared__ float bv[64]; __shared__ int bi[64];
    __shared__ float par[6];
    feat[t] = (float)hb[n * 64 + t];
    __syncthreads();
    if (t < 18) {
        int head = t / 6, cls = t % 6;
        const float* w = (head == 0 ? sxw : head == 1 ? syw : thw) + cls * 64;
        const float* b = (head == 0 ? sxb : head == 1 ? syb : thb);
        float acc = b[cls];
        for (int c = 0; c < 64; ++c) acc += feat[c] * w[c];
        lin[head][cls] = acc;
    }
    float best = NEG_BIG; int bidx = 0x7fffffff;
    for (int p = t; p < 441; p += 64) {
        float v = d2[n * 441 + p];
        if (v > best) { best = v; bidx = p; }
    }
    bv[t] = best; bi[t] = bidx;
    __syncthreads();
    if (t == 0) {
        float b0 = bv[0]; int i0 = bi[0];
        for (int k = 1; k < 64; ++k)
            if (bv[k] > b0 || (bv[k] == b0 && bi[k] < i0)) { b0 = bv[k]; i0 = bi[k]; }
        int ax[3];
        for (int hd = 0; hd < 3; ++hd) {
            float mm = lin[hd][0]; int a = 0;
            for (int k2 = 1; k2 < 6; ++k2) if (lin[hd][k2] > mm) { mm = lin[hd][k2]; a = k2; }
            ax[hd] = a;
        }
        float sx = ax[0] / 10.f + 0.1f;      // (argmax/5)/2 + 0.1
        float sy = ax[1] / 10.f + 0.1f;
        float th = ax[2] * (3.14159265358979323846f / 6.f);
        float c = cosf(th), s = sinf(th);
        float S11 = c * c * sx + s * s * sy;
        float S22 = s * s * sx + c * c * sy;
        float S12 = c * s * (sx - sy);
        float det = sx * sy;
        par[0] = S22 / det; par[1] = -S12 / det; par[2] = S11 / det;
        par[3] = (i0 / 21 - 10) * 0.1f;
        par[4] = (i0 % 21 - 10) * 0.1f;
        par[5] = 1.f / (2.f * 3.14159265358979323846f * sqrtf(det));
    }
    __syncthreads();
    float ia = par[0], ibv = par[1], ic = par[2];
    float mi0 = par[3], mi1 = par[4], invf = par[5];
    for (int p = t; p < 441; p += 64) {
        int a = p / 21, b2 = p % 21;
        float dx = (a - 10) * 0.1f - mi0;
        float dy = (b2 - 10) * 0.1f - mi1;
        float q = ia * dx * dx + 2.f * ibv * dx * dy + ic * dy * dy;
        float m = expf(-0.5f * q) * invf;
        scl[n * 441 + p] = m / nrm[n * 441 + p];
    }
}

// ---------------------------------------------------------------------------
// Fused masked cosine-similarity (WMMA) + per-row top-3.  One wave per
// (pair, 16-row p-tile).  WMMA operands loaded DIRECTLY from global memory
// (2x b128 per lane per operand, per the ISA 16-bit operand layout) -- no
// LDS staging for A/B at all.  Deterministic per-tile partials.
// ---------------------------------------------------------------------------
__global__ __launch_bounds__(32)
void sim_topk(const _Float16* __restrict__ fT, const float* __restrict__ scl,
              float* __restrict__ partial) {
    const int pt = blockIdx.x;                 // p-tile 0..27
    const int pr = blockIdx.y;                 // pair 0..374
    const int ni = pr / 5, nj = 75 + pr % 5;
    const int lane = threadIdx.x;
    const int lhi = lane >> 4, lrow = lane & 15;
    __shared__ float St[16][17];
    __shared__ float s1[16], s2[16];
    __shared__ float mtop[16][3];
    __shared__ float rsum[16];
    const int p0 = pt * 16;
    const _Float16* fA = fT + (size_t)ni * 441 * 64;
    const _Float16* fB = fT + (size_t)nj * 441 * 64;
    const int kb = lhi ? 8 : 0;
    // A operand: lane supplies row p0+lrow, K-half per lhi.
    const int pA = p0 + lrow;
    const bool pv = pA < 441;
    const _Float16* rowA = fA + (size_t)(pv ? pA : 0) * 64;
    v16h a0 = pv ? ldpair(rowA, kb)      : zero16();
    v16h a1 = pv ? ldpair(rowA + 32, kb) : zero16();
    if (lane < 16) {
        int p = p0 + lane;
        s1[lane] = (p < 441) ? scl[ni * 441 + p] : 0.f;
    }
    float t0 = NEG_BIG, t1 = NEG_BIG, t2 = NEG_BIG;
    const int col0 = lhi * 8;
    for (int qt = 0; qt < 28; ++qt) {
        int q0 = qt * 16;
        int q = q0 + lrow;
        bool qv = q < 441;
        const _Float16* rowB = fB + (size_t)(qv ? q : 0) * 64;
        v16h b0 = qv ? ldpair(rowB, kb)      : zero16();
        v16h b1 = qv ? ldpair(rowB + 32, kb) : zero16();
        if (lane < 16) {
            int qq = q0 + lane;
            s2[lane] = (qq < 441) ? scl[nj * 441 + qq] : 0.f;
        }
        // prefetch next q-tile rows (global_prefetch_b8)
        int qn = q0 + 16 + lrow;
        if (qn < 441) __builtin_prefetch(fB + (size_t)qn * 64, 0, 1);
        __syncthreads();
        v8f c = {};
        c = __builtin_amdgcn_wmma_f32_16x16x32_f16(false, a0, false, b0, (short)0, c, false, false);
        c = __builtin_amdgcn_wmma_f32_16x16x32_f16(false, a1, false, b1, (short)0, c, false, false);
#pragma unroll
        for (int r = 0; r < 8; ++r) {
            int m = r + 8 * lhi;
            float v = c[r] * s1[m] * s2[lrow];
            St[m][lrow] = qv ? v : NEG_BIG;    // exclude padded q columns
        }
        __syncthreads();
#pragma unroll
        for (int cc = 0; cc < 8; ++cc) {
            float v = St[lrow][col0 + cc];
            if (v > t0)      { t2 = t1; t1 = t0; t0 = v; }
            else if (v > t1) { t2 = t1; t1 = v; }
            else if (v > t2) { t2 = v; }
        }
    }
    if (lhi) { mtop[lrow][0] = t0; mtop[lrow][1] = t1; mtop[lrow][2] = t2; }
    __syncthreads();
    if (!lhi) {
        for (int e = 0; e < 3; ++e) {
            float v = mtop[lrow][e];
            if (v > t0)      { t2 = t1; t1 = t0; t0 = v; }
            else if (v > t1) { t2 = t1; t1 = v; }
            else if (v > t2) { t2 = v; }
        }
        int p = p0 + lrow;
        rsum[lrow] = (p < 441) ? (t0 + t1 + t2) : 0.f;
    }
    __syncthreads();
    if (lane == 0) {
        float s = 0.f;
        for (int r = 0; r < 16; ++r) s += rsum[r];
        partial[pr * 28 + pt] = s;
    }
}

// ---------------------------------------------------------------------------
// Reduce partials -> sims, 5x5 classifier, softmax
// ---------------------------------------------------------------------------
__global__ __launch_bounds__(512)
void finalize(const float* __restrict__ partial, const float* __restrict__ scw,
              const float* __restrict__ scb, float* __restrict__ out) {
    __shared__ float sims[375];
    int t = threadIdx.x;
    for (int s = t; s < 375; s += 512) {
        float acc = 0.f;
        for (int pt = 0; pt < 28; ++pt) acc += partial[s * 28 + pt];
        sims[s] = acc;
    }
    __syncthreads();
    if (t < 75) {
        float lg[5]; float mx = NEG_BIG;
        for (int k = 0; k < 5; ++k) {
            float acc = scb[k];
            for (int j = 0; j < 5; ++j) acc += sims[t * 5 + j] * scw[k * 5 + j];
            lg[k] = acc; mx = fmaxf(mx, acc);
        }
        float den = 0.f;
        for (int k = 0; k < 5; ++k) { lg[k] = expf(lg[k] - mx); den += lg[k]; }
        for (int k = 0; k < 5; ++k) out[t * 5 + k] = lg[k] / den;
    }
}

// ---------------------------------------------------------------------------
// Host orchestration
// ---------------------------------------------------------------------------
extern "C" void kernel_launch(void* const* d_in, const int* in_sizes, int n_in,
                              void* d_out, int out_size, void* d_ws, size_t ws_size,
                              hipStream_t stream) {
    (void)in_sizes; (void)n_in; (void)out_size; (void)ws_size;
    // input index map (setup_inputs dict order, bn tuples as g,b,m,v)
    enum { I1 = 0, I2, FW0, FBN0, FW1 = FBN0 + 4, FBN1, FW2 = FBN1 + 4, FBN2,
           FW3 = FBN2 + 4, FBN3, GW0 = FBN3 + 4, GBN0, GW1 = GBN0 + 4, GBN1,
           GW2 = GBN1 + 4, GB2, DW0, DBN0, DW1 = DBN0 + 4, DBN1, DW2 = DBN1 + 4,
           DB2, SXW, SXB, SYW, SYB, THW, THB, SCW, SCB };
    auto F = [&](int i) { return (const float*)d_in[i]; };

    char* ws = (char*)d_ws;
    size_t off = 0;
    auto alloc = [&](size_t bytes) {
        size_t p = off; off += (bytes + 255) & ~(size_t)255; return p;
    };
    const size_t P42 = 1764, P21 = 441;
    _Float16* y1    = (_Float16*)(ws + alloc(NIMG * 64 * P42 * 2));  // pooled conv0
    _Float16* y2a   = (_Float16*)(ws + alloc(NIMG * 64 * P42 * 2));  // conv1 pre-pool
    _Float16* y2    = (_Float16*)(ws + alloc(NIMG * 64 * P21 * 2));
    _Float16* y3    = (_Float16*)(ws + alloc(NIMG * 64 * P21 * 2));
    _Float16* f     = (_Float16*)(ws + alloc(NIMG * 64 * P21 * 2));
    _Float16* fT    = (_Float16*)(ws + alloc(NIMG * P21 * 64 * 2));
    _Float16* g0p   = (_Float16*)(ws + alloc(NIMG * 64 * 49 * 2));
    _Float16* g1b   = (_Float16*)(ws + alloc(NIMG * 64 * 9 * 2));
    _Float16* hb    = (_Float16*)(ws + alloc(NIMG * 64 * 2));
    float* d0       = (float*)(ws + alloc(NIMG * 64 * 16 * 4));
    float* d1       = (float*)(ws + alloc(NIMG * 64 * 100 * 4));
    float* d2       = (float*)(ws + alloc(NIMG * P21 * 4));
    float* nrm      = (float*)(ws + alloc(NIMG * P21 * 4));
    float* scl      = (float*)(ws + alloc(NIMG * P21 * 4));
    _Float16* wk0   = (_Float16*)(ws + alloc(32 * 64 * 2));
    _Float16* wkc[6]; // f1,f2,f3,g0,g1,g2
    for (int i = 0; i < 6; ++i) wkc[i] = (_Float16*)(ws + alloc(576 * 64 * 2));
    float* bnb      = (float*)(ws + alloc(9 * 2 * 64 * 4)); // 9 (scale,shift) pairs
    float* partial  = (float*)(ws + alloc(375 * 28 * 4));
    _Float16* g0b   = y1;  // reuse: y1 dead after conv1

    auto bnS = [&](int slot) { return bnb + slot * 128; };
    auto bnH = [&](int slot) { return bnb + slot * 128 + 64; };

    // ---- prep ----
    prep_wk<<<(32 * 64 + 255) / 256, 256, 0, stream>>>(F(FW0), wk0, 3, 32);
    const int wsrc[6] = { FW1, FW2, FW3, GW0, GW1, GW2 };
    for (int i = 0; i < 6; ++i)
        prep_wk<<<(576 * 64 + 255) / 256, 256, 0, stream>>>(F(wsrc[i]), wkc[i], 64, 576);
    const int bnsrc[8] = { FBN0, FBN1, FBN2, FBN3, GBN0, GBN1, DBN0, DBN1 };
    const int bnslot[8] = { 0, 1, 2, 3, 4, 5, 7, 8 };
    for (int i = 0; i < 8; ++i)
        prep_bn<<<1, 64, 0, stream>>>(F(bnsrc[i]), F(bnsrc[i] + 1), F(bnsrc[i] + 2),
                                      F(bnsrc[i] + 3), bnS(bnslot[i]), bnH(bnslot[i]), 64);
    prep_bias<<<1, 64, 0, stream>>>(F(GB2), bnS(6), bnH(6), 64);

    // ---- feature extractor ----
    {
        int total = NIMG * 64 * 42 * 42;
        conv0_pool<<<(total + 255) / 256, 256, 0, stream>>>(F(I1), F(I2), F(FW0),
                                                            bnS(0), bnH(0), y1);
    }
    auto conv = [&](const _Float16* x, const _Float16* wk, int slot, _Float16* y,
                    _Float16* yT, int IH, int IW, int OH, int OW, int stride,
                    int pad, int nk, int Kreal) {
        dim3 grid((OH * OW + 15) / 16, NIMG);
        conv3x3_wmma<<<grid, 128, 0, stream>>>(x, wk, bnS(slot), bnH(slot), y, yT,
                                               IH, IW, OH, OW, 64, stride, pad, nk, Kreal);
    };
    conv(y1, wkc[0], 1, y2a, nullptr, 42, 42, 42, 42, 1, 1, 18, 576);   // f_w1
    {
        int total = NIMG * 64 * 21 * 21;
        maxpool<<<(total + 255) / 256, 256, 0, stream>>>(y2a, y2, NIMG * 64, 42, 42, 2, 2, 21, 21);
    }
    conv(y2, wkc[1], 2, y3, nullptr, 21, 21, 21, 21, 1, 1, 18, 576);    // f_w2
    conv(y3, wkc[2], 3, f, fT, 21, 21, 21, 21, 1, 1, 18, 576);          // f_w3 (+transposed)
    {
        int total = NIMG * 441;
        norm_kernel<<<(total + 255) / 256, 256, 0, stream>>>(fT, nrm);
    }

    // ---- mask path ----
    conv(f, wkc[3], 4, g0b, nullptr, 21, 21, 21, 21, 1, 1, 18, 576);    // g_w0
    {
        int total = NIMG * 64 * 49;
        maxpool<<<(total + 255) / 256, 256, 0, stream>>>(g0b, g0p, NIMG * 64, 21, 21, 3, 3, 7, 7);
    }
    conv(g0p, wkc[4], 5, g1b, nullptr, 7, 7, 3, 3, 2, 0, 18, 576);      // g_w1 (stride 2)
    conv(g1b, wkc[5], 6, hb, nullptr, 3, 3, 1, 1, 1, 0, 18, 576);       // g_w2 (bias+lrelu)
    deconv0<<<(NIMG * 64 * 16 + 255) / 256, 256, 0, stream>>>(hb, F(DW0), bnS(7), bnH(7), d0);
    deconv1<<<(NIMG * 64 * 100 + 255) / 256, 256, 0, stream>>>(d0, F(DW1), bnS(8), bnH(8), d1);
    deconv2<<<(NIMG * 441 + 255) / 256, 256, 0, stream>>>(d1, F(DW2), F(DB2), d2);
    mask_kernel<<<NIMG, 64, 0, stream>>>(hb, d2, nrm, F(SXW), F(SXB), F(SYW), F(SYB),
                                         F(THW), F(THB), scl);

    // ---- pairwise masked cosine similarity + top-3 ----
    {
        dim3 grid(28, 375);
        sim_topk<<<grid, 32, 0, stream>>>(fT, scl, partial);
    }
    finalize<<<1, 512, 0, stream>>>(partial, F(SCW), F(SCB), (float*)d_out);
}
